// GPT_57947698758351
// MI455X (gfx1250) — matrix operand order repaired and checked by reading, not compiled, not run on hardware
//
#include <hip/hip_runtime.h>

// GPT-2 small forward on gfx1250 (CDNA5), wave32, WMMA bf16 path.
// B=2 T=1024 E=768 H=12 HS=64 L=12 F4=3072 V=50304, rows M = B*T = 2048.

typedef __attribute__((ext_vector_type(16))) __bf16 v16bf;
typedef __attribute__((ext_vector_type(2)))  __bf16 v2bf;
typedef __attribute__((ext_vector_type(8)))  float  v8f;

#define GPT_T  1024
#define GPT_E  768
#define GPT_F4 3072
#define GPT_V  50304
#define GPT_M  2048

// ---- async global->LDS staging (CDNA5), graceful fallback ----------------
#if __has_builtin(__builtin_amdgcn_global_load_async_to_lds_b128) && \
    __has_builtin(__builtin_amdgcn_s_wait_asynccnt)
#define GPT_ASYNC 1
#else
#define GPT_ASYNC 0
#endif

#if GPT_ASYNC
typedef int b128_t __attribute__((vector_size(4 * sizeof(int))));
typedef __attribute__((address_space(1))) b128_t* gptr128;
typedef __attribute__((address_space(3))) b128_t* lptr128;
#endif

__device__ __forceinline__ void cp_b128(void* lds, const void* g) {
#if GPT_ASYNC
  __builtin_amdgcn_global_load_async_to_lds_b128((gptr128)g, (lptr128)lds,
                                                 0, 0);
#else
  *(float4*)lds = *(const float4*)g;
#endif
}
__device__ __forceinline__ void cp_wait() {
#if GPT_ASYNC
  __builtin_amdgcn_s_wait_asynccnt(0);
#endif
}

// ---- Tensor Data Mover staging (CDNA5), graceful fallback ----------------
#if __has_builtin(__builtin_amdgcn_tensor_load_to_lds) && \
    __has_builtin(__builtin_amdgcn_s_wait_tensorcnt)
#define GPT_TDM 1
#else
#define GPT_TDM 0
#endif

#if GPT_TDM
typedef unsigned int v4u __attribute__((ext_vector_type(4)));
typedef int v8i_t __attribute__((ext_vector_type(8)));
typedef int v4i_t __attribute__((ext_vector_type(4)));
typedef __attribute__((address_space(3))) void* lds_vp;

__device__ __forceinline__ unsigned lds_off(void* p) {
  return (unsigned)(unsigned long long)(lds_vp)p;  // AS3 ptr == LDS byte off
}

// 2D tile DMA: bf16 elements, tile (tile_w x tile_h), row stride = stride.
// D# layout per CDNA5 ISA ch.8 (group0 128b, group1 256b, groups2/3 unused).
__device__ __forceinline__ void tdm_load_2d_bf16(const void* gsrc,
                                                 unsigned ldsaddr, int tile_w,
                                                 int tile_h, int stride) {
  const unsigned long long ga = (unsigned long long)gsrc;
  v4u g0;
  g0.x = 1u;                                  // count=1, user descriptor
  g0.y = ldsaddr;                             // lds_addr [63:32]
  g0.z = (unsigned)(ga & 0xFFFFFFFFu);        // global_addr lo
  g0.w = (unsigned)((ga >> 32) & 0x1FFFFFFu)  // global_addr hi (57-bit)
         | (2u << 30);                        // type = 2 ("image")
  v8i_t g1;
  g1[0] = 1 << 16;                    // wg_mask=0, data_size=1 (2 bytes)
  g1[1] = (int)0xFFFF0000u;           // tensor_dim0 lo16 (dim0 = 0x7FFFFFFF)
  g1[2] = (int)0xFFFF7FFFu;           // dim0 hi16 | tensor_dim1 lo16
  g1[3] = (tile_w << 16) | 0x7FFF;    // tensor_dim1 hi16 | tile_dim0
  g1[4] = tile_h & 0xFFFF;            // tile_dim1 | tile_dim2=0
  g1[5] = stride;                     // tensor_dim0_stride [31:0]
  g1[6] = 0;                          // stride hi | tensor_dim1_stride lo
  g1[7] = 0;
  const v4i_t gz4 = {0, 0, 0, 0};
  const v8i_t gz8 = {0, 0, 0, 0, 0, 0, 0, 0};
  __builtin_amdgcn_tensor_load_to_lds(g0, g1, gz4, gz4, gz8, 0);
}
#endif

__device__ __forceinline__ v8f wmma_bf16(v16bf a, v16bf b, v8f c) {
  // D = A(16x32 bf16) * B(32x16 bf16) + C(16x16 f32)
  return __builtin_amdgcn_wmma_f32_16x16x32_bf16(false, a, false, b,
                                                 (short)0, c, false, false);
}

// A-operand fragment: 16x32 bf16 tile from LDS, row-major [row][k], stride ld.
__device__ __forceinline__ v16bf load_a_frag(const __bf16* s, int ld, int row0,
                                             int k0, int lane) {
  const int row   = row0 + (lane & 15);
  const int khalf = (lane >> 4) << 4;
  const __bf16* p = s + row * ld + k0;
  v16bf a;
#pragma unroll
  for (int v = 0; v < 8; ++v) {
    const int kk = khalf + (v << 1);
    a[2 * v]     = p[kk];
    a[2 * v + 1] = p[kk + 1];
  }
  return a;
}

// B-operand fragment where LDS holds B transposed: s[n][k] (stride ld in k).
__device__ __forceinline__ v16bf load_b_fragT(const __bf16* s, int ld, int col0,
                                              int k0, int lane) {
  const int col = col0 + (lane & 15);
  const int kb  = (lane >> 4) << 4;
  const __bf16* p = s + col * ld + k0 + kb;
  v16bf b;
#pragma unroll
  for (int v = 0; v < 8; ++v) {
    b[2 * v]     = p[2 * v];
    b[2 * v + 1] = p[2 * v + 1];
  }
  return b;
}

// B-operand fragment where LDS holds B as [k][n] row-major (stride ld in n).
__device__ __forceinline__ v16bf load_b_fragKN(const __bf16* s, int ld, int col0,
                                               int k0, int lane) {
  const int col = col0 + (lane & 15);
  const int kb  = (lane >> 4) << 4;
  const __bf16* p = s + (k0 + kb) * ld + col;
  v16bf b;
#pragma unroll
  for (int v = 0; v < 8; ++v) {
    b[2 * v]     = p[(2 * v) * ld];
    b[2 * v + 1] = p[(2 * v + 1) * ld];
  }
  return b;
}

// ---------------------------------------------------------------- embedding
__global__ __launch_bounds__(256) void embed_k(const int* __restrict__ idx,
                                               const float* __restrict__ tok,
                                               const float* __restrict__ pos,
                                               float* __restrict__ x) {
  const int r  = blockIdx.x;  // 0..2047
  const int t  = r & (GPT_T - 1);
  const int tk = idx[r];
  for (int i = threadIdx.x; i < GPT_E; i += 256)
    x[(size_t)r * GPT_E + i] =
        tok[(size_t)tk * GPT_E + i] + pos[(size_t)t * GPT_E + i];
}

// ---------------------------------------------------------------- layernorm
__global__ __launch_bounds__(256) void layernorm_k(const float* __restrict__ x,
                                                   const float* __restrict__ g,
                                                   const float* __restrict__ bt,
                                                   __bf16* __restrict__ out) {
  const int row = blockIdx.x;
  const int tid = threadIdx.x;
  const float* xr = x + (size_t)row * GPT_E;
  float s = 0.f, s2 = 0.f;
#pragma unroll
  for (int it = 0; it < 3; ++it) {  // 3*256 = 768
    const float v = xr[tid + it * 256];
    s += v;
    s2 += v * v;
  }
#pragma unroll
  for (int m = 16; m >= 1; m >>= 1) {
    s  += __shfl_xor(s, m, 32);
    s2 += __shfl_xor(s2, m, 32);
  }
  __shared__ float sh0[8], sh1[8];
  const int wave = tid >> 5, lane = tid & 31;
  if (lane == 0) { sh0[wave] = s; sh1[wave] = s2; }
  __syncthreads();
  float ts = 0.f, ts2 = 0.f;
#pragma unroll
  for (int i = 0; i < 8; ++i) { ts += sh0[i]; ts2 += sh1[i]; }
  const float mean = ts * (1.f / GPT_E);
  const float var  = ts2 * (1.f / GPT_E) - mean * mean;
  const float rstd = rsqrtf(var + 1e-5f);
#pragma unroll
  for (int it = 0; it < 3; ++it) {
    const int i = tid + it * 256;
    out[(size_t)row * GPT_E + i] =
        (__bf16)((xr[i] - mean) * rstd * g[i] + bt[i]);
  }
}

// ------------------------------------------------------------------- GEMM
// C[M,N] = A[M,K](bf16) * B[K,N](f32 weights -> bf16 on load)
// tile 128x128x64, 256 threads = 8 waves (4 x 2); wave owns 32x64 (2x4 WMMAs)
template <bool HAS_BIAS, bool RELU, bool HAS_RES, bool OUT_F32>
__device__ __forceinline__ void gemm_body(
    const __bf16* __restrict__ A, int lda,
    const float* __restrict__ Bw, int ldb,
    const float* __restrict__ bias,
    const float* __restrict__ resid,
    float* __restrict__ outF,
    __bf16* __restrict__ outB,
    int N, int K, int bm, int bn,
    __bf16* As, __bf16* Bs) {
  const int tid  = threadIdx.x;
  const int lane = tid & 31, wave = tid >> 5;
  const int wm = wave >> 1, wn = wave & 1;
#if GPT_TDM
  const int wv0 = __builtin_amdgcn_readfirstlane(tid >> 5);  // scalar wave id
  const unsigned asOff = lds_off(As);
#endif

  const v8f zero8 = {0.f, 0.f, 0.f, 0.f, 0.f, 0.f, 0.f, 0.f};
  v8f acc[2][4];
#pragma unroll
  for (int i = 0; i < 2; ++i)
#pragma unroll
    for (int j = 0; j < 4; ++j) acc[i][j] = zero8;

  for (int k0 = 0; k0 < K; k0 += 64) {
    if (k0 + 64 < K) {  // prefetch next K-slice (uniform branch, EXEC intact)
      __builtin_prefetch(&Bw[(size_t)(k0 + 64 + (tid >> 5)) * ldb + bn +
                             ((tid & 31) << 2)], 0, 0);
    }
    // stage A: 128x64 bf16 tile; TDM descriptor DMA (wave 0) or async copies
#if GPT_TDM
    if (wv0 == 0)
      tdm_load_2d_bf16(&A[(size_t)bm * lda + k0], asOff, 64, 128, lda);
#else
#pragma unroll
    for (int it = 0; it < 4; ++it) {
      const int i = (tid + it * 256) * 8;
      const int r = i >> 6, c = i & 63;
      cp_b128(&As[r * 64 + c], &A[(size_t)(bm + r) * lda + k0 + c]);
    }
#endif
    // stage B: two f32 k-rows per item, pack bf16 (k,k+1) pairs -> b32 store
#pragma unroll
    for (int it = 0; it < 4; ++it) {
      const int p  = tid + it * 256;     // 0..1023
      const int n4 = (p & 31) << 2;      // 0..124
      const int kk = (p >> 5) << 1;      // 0..62 (even)
      const float* bp = &Bw[(size_t)(k0 + kk) * ldb + bn + n4];
      const float4 va = *(const float4*)bp;
      const float4 vb = *(const float4*)(bp + ldb);
      v2bf t0; t0.x = (__bf16)va.x; t0.y = (__bf16)vb.x;
      v2bf t1; t1.x = (__bf16)va.y; t1.y = (__bf16)vb.y;
      v2bf t2; t2.x = (__bf16)va.z; t2.y = (__bf16)vb.z;
      v2bf t3; t3.x = (__bf16)va.w; t3.y = (__bf16)vb.w;
      *(v2bf*)&Bs[(n4 + 0) * 64 + kk] = t0;
      *(v2bf*)&Bs[(n4 + 1) * 64 + kk] = t1;
      *(v2bf*)&Bs[(n4 + 2) * 64 + kk] = t2;
      *(v2bf*)&Bs[(n4 + 3) * 64 + kk] = t3;
    }
#if GPT_TDM
    if (wv0 == 0) __builtin_amdgcn_s_wait_tensorcnt(0);
#else
    cp_wait();
#endif
    __syncthreads();
#pragma unroll
    for (int ks = 0; ks < 64; ks += 32) {
      v16bf af0 = load_a_frag(As, 64, wm * 32, ks, lane);
      v16bf af1 = load_a_frag(As, 64, wm * 32 + 16, ks, lane);
#pragma unroll
      for (int nt = 0; nt < 4; ++nt) {
        v16bf bf = load_b_fragT(Bs, 64, wn * 64 + nt * 16, ks, lane);
        acc[0][nt] = wmma_bf16(af0, bf, acc[0][nt]);
        acc[1][nt] = wmma_bf16(af1, bf, acc[1][nt]);
      }
    }
    __syncthreads();
  }

  const int half8 = (lane >> 4) << 3;
  float bvals[4];
  if (HAS_BIAS) {
#pragma unroll
    for (int nt = 0; nt < 4; ++nt)
      bvals[nt] = bias[bn + wn * 64 + nt * 16 + (lane & 15)];
  }
#pragma unroll
  for (int mt = 0; mt < 2; ++mt)
#pragma unroll
    for (int nt = 0; nt < 4; ++nt) {
      const int r0 = bm + wm * 32 + mt * 16 + half8;
      const int c  = bn + wn * 64 + nt * 16 + (lane & 15);
#pragma unroll
      for (int vv = 0; vv < 8; ++vv) {
        float val = acc[mt][nt][vv];
        if (HAS_BIAS) val += bvals[nt];
        if (RELU) val = fmaxf(val, 0.f);
        const size_t o = (size_t)(r0 + vv) * N + c;
        if (HAS_RES) val += resid[o];
        if (OUT_F32) outF[o] = val;
        else         outB[o] = (__bf16)val;
      }
    }
}

template <bool HAS_BIAS, bool RELU, bool HAS_RES, bool OUT_F32>
__global__ __launch_bounds__(256) void gemm_k(
    const __bf16* __restrict__ A, int lda,
    const float* __restrict__ Bw, int ldb,
    const float* __restrict__ bias,
    const float* __restrict__ resid,
    float* __restrict__ outF,
    __bf16* __restrict__ outB,
    int N, int K) {
  __shared__ __align__(16) __bf16 As[128 * 64];  // [row][k]      16 KB
  __shared__ __align__(16) __bf16 Bs[128 * 64];  // [n][k] transp 16 KB
  gemm_body<HAS_BIAS, RELU, HAS_RES, OUT_F32>(
      A, lda, Bw, ldb, bias, resid, outF, outB, N, K,
      blockIdx.y * 128, blockIdx.x * 128, As, Bs);
}

// Fused QKV projection: gridDim.z = 3 selects {wq->q, wk->k, wv->v}.
__global__ __launch_bounds__(256) void gemm_qkv_k(
    const __bf16* __restrict__ A,
    const float* __restrict__ w0, const float* __restrict__ w1,
    const float* __restrict__ w2,
    __bf16* __restrict__ o0, __bf16* __restrict__ o1,
    __bf16* __restrict__ o2) {
  __shared__ __align__(16) __bf16 As[128 * 64];
  __shared__ __align__(16) __bf16 Bs[128 * 64];
  const int z = blockIdx.z;  // uniform scalar select
  const float* Bw = (z == 0) ? w0 : (z == 1) ? w1 : w2;
  __bf16*      ob = (z == 0) ? o0 : (z == 1) ? o1 : o2;
  gemm_body<false, false, false, false>(
      A, GPT_E, Bw, GPT_E, nullptr, nullptr, nullptr, ob, GPT_E, GPT_E,
      blockIdx.y * 128, blockIdx.x * 128, As, Bs);
}

// ------------------------------------------------------- flash attention
// grid (T/64, H, B), 128 threads = 4 waves; wave w owns q rows [w*16, w*16+16)
__global__ __launch_bounds__(128) void attn_k(const __bf16* __restrict__ qp,
                                              const __bf16* __restrict__ kp,
                                              const __bf16* __restrict__ vp,
                                              __bf16* __restrict__ op) {
  __shared__ __align__(16) __bf16 Qs[64 * 64];
  __shared__ __align__(16) __bf16 Ks[64 * 64];
  __shared__ __align__(16) __bf16 Vs[64 * 64];
  __shared__ __align__(16) __bf16 Ps[64 * 64];
  const int tid = threadIdx.x, lane = tid & 31, w = tid >> 5;
  const int qblk = blockIdx.x, h = blockIdx.y, b = blockIdx.z;
  const size_t base = (size_t)b * GPT_T * GPT_E + h * 64;

#pragma unroll
  for (int it = 0; it < 4; ++it) {  // 64x64 bf16 = 512 b128 chunks
    const int i = (tid + it * 128) * 8;
    const int r = i >> 6, c = i & 63;
    cp_b128(&Qs[r * 64 + c], &qp[base + (size_t)(qblk * 64 + r) * GPT_E + c]);
  }

  const v8f zero8 = {0.f, 0.f, 0.f, 0.f, 0.f, 0.f, 0.f, 0.f};
  v8f oacc[4];
  float mrun[8], lrun[8];
#pragma unroll
  for (int t = 0; t < 4; ++t) oacc[t] = zero8;
#pragma unroll
  for (int v = 0; v < 8; ++v) { mrun[v] = -1e30f; lrun[v] = 0.f; }
  const int half8 = (lane >> 4) << 3;

  for (int j = 0; j <= qblk; ++j) {
    __syncthreads();  // prior P*V done before K/V overwrite
#pragma unroll
    for (int it = 0; it < 4; ++it) {
      const int i = (tid + it * 128) * 8;
      const int r = i >> 6, c = i & 63;
      const size_t g = base + (size_t)(j * 64 + r) * GPT_E + c;
      cp_b128(&Ks[r * 64 + c], &kp[g]);
      cp_b128(&Vs[r * 64 + c], &vp[g]);
    }
    cp_wait();
    __syncthreads();

    // S = Q * K^T  (64 q-rows x 64 kv-cols, contraction over HS=64)
    v8f s[4];
#pragma unroll
    for (int nt = 0; nt < 4; ++nt) s[nt] = zero8;
#pragma unroll
    for (int ks = 0; ks < 64; ks += 32) {
      v16bf a = load_a_frag(Qs, 64, w * 16, ks, lane);
#pragma unroll
      for (int nt = 0; nt < 4; ++nt)
        s[nt] = wmma_bf16(a, load_b_fragT(Ks, 64, nt * 16, ks, lane), s[nt]);
    }

    const int rowbase = qblk * 64 + w * 16 + half8;
    const int colb    = j * 64 + (lane & 15);
#pragma unroll
    for (int vv = 0; vv < 8; ++vv) {
      const int r = rowbase + vv;
      float val[4];
      float rmax = -1e30f;
#pragma unroll
      for (int nt = 0; nt < 4; ++nt) {
        float xv = s[nt][vv] * 0.125f;        // HS^-0.5
        if (colb + nt * 16 > r) xv = -1e30f;  // causal (if-converted cndmask)
        val[nt] = xv;
        rmax = fmaxf(rmax, xv);
      }
#pragma unroll
      for (int m = 8; m >= 1; m >>= 1)
        rmax = fmaxf(rmax, __shfl_xor(rmax, m, 32));
      const float mnew  = fmaxf(mrun[vv], rmax);
      const float alpha = __expf(mrun[vv] - mnew);
      float rsum = 0.f;
#pragma unroll
      for (int nt = 0; nt < 4; ++nt) {
        const float p = __expf(val[nt] - mnew);
        rsum += p;
        Ps[(w * 16 + half8 + vv) * 64 + nt * 16 + (lane & 15)] = (__bf16)p;
      }
#pragma unroll
      for (int m = 8; m >= 1; m >>= 1) rsum += __shfl_xor(rsum, m, 32);
      lrun[vv] = lrun[vv] * alpha + rsum;
      mrun[vv] = mnew;
#pragma unroll
      for (int ot = 0; ot < 4; ++ot) oacc[ot][vv] *= alpha;
    }
    __syncthreads();

    // O += P * V (contraction over kv tile of 64)
#pragma unroll
    for (int ks = 0; ks < 64; ks += 32) {
      v16bf a = load_a_frag(Ps, 64, w * 16, ks, lane);
#pragma unroll
      for (int ot = 0; ot < 4; ++ot)
        oacc[ot] = wmma_bf16(a, load_b_fragKN(Vs, 64, ot * 16, ks, lane),
                             oacc[ot]);
    }
  }

#pragma unroll
  for (int vv = 0; vv < 8; ++vv) {
    const float inv = 1.f / lrun[vv];
    const int r = qblk * 64 + w * 16 + half8 + vv;
#pragma unroll
    for (int ot = 0; ot < 4; ++ot) {
      const int c = ot * 16 + (lane & 15);
      op[base + (size_t)r * GPT_E + c] = (__bf16)(oacc[ot][vv] * inv);
    }
  }
}

// ------------------------------------------------------------------ host
extern "C" void kernel_launch(void* const* d_in, const int* in_sizes, int n_in,
                              void* d_out, int out_size, void* d_ws,
                              size_t ws_size, hipStream_t stream) {
  const int*   idx   = (const int*)d_in[0];
  const float* tok   = (const float*)d_in[1];
  const float* pos   = (const float*)d_in[2];
  const float* wq    = (const float*)d_in[3];
  const float* wk    = (const float*)d_in[4];
  const float* wv    = (const float*)d_in[5];
  const float* wproj = (const float*)d_in[6];
  const float* bproj = (const float*)d_in[7];
  const float* ln1g  = (const float*)d_in[8];
  const float* ln1b  = (const float*)d_in[9];
  const float* ln2g  = (const float*)d_in[10];
  const float* ln2b  = (const float*)d_in[11];
  const float* wfc   = (const float*)d_in[12];
  const float* bfc   = (const float*)d_in[13];
  const float* wpr2  = (const float*)d_in[14];
  const float* bpr2  = (const float*)d_in[15];
  const float* lnfg  = (const float*)d_in[16];
  const float* lnfb  = (const float*)d_in[17];
  const float* lmw   = (const float*)d_in[18];
  const float* lmb   = (const float*)d_in[19];

  char* w = (char*)d_ws;
  float*  x    = (float*)w;  w += (size_t)GPT_M * GPT_E * sizeof(float);
  __bf16* xn   = (__bf16*)w; w += (size_t)GPT_M * GPT_E * sizeof(__bf16);
  __bf16* qb   = (__bf16*)w; w += (size_t)GPT_M * GPT_E * sizeof(__bf16);
  __bf16* kb   = (__bf16*)w; w += (size_t)GPT_M * GPT_E * sizeof(__bf16);
  __bf16* vb   = (__bf16*)w; w += (size_t)GPT_M * GPT_E * sizeof(__bf16);
  __bf16* attb = (__bf16*)w; w += (size_t)GPT_M * GPT_E * sizeof(__bf16);
  __bf16* hb   = (__bf16*)w; w += (size_t)GPT_M * GPT_F4 * sizeof(__bf16);

  embed_k<<<GPT_M, 256, 0, stream>>>(idx, tok, pos, x);

  const dim3 gEE(GPT_E / 128, GPT_M / 128);      // (6,16)
  const dim3 gQKV(GPT_E / 128, GPT_M / 128, 3);  // (6,16,3)
  const dim3 gEF(GPT_F4 / 128, GPT_M / 128);     // (24,16)
  const dim3 gEV(GPT_V / 128, GPT_M / 128);      // (393,16)
  const dim3 gATT(GPT_T / 64, 12, 2);

  for (int l = 0; l < 12; ++l) {
    const size_t oEE = (size_t)l * GPT_E * GPT_E;
    const size_t oEF = (size_t)l * GPT_E * GPT_F4;
    layernorm_k<<<GPT_M, 256, 0, stream>>>(x, ln1g + l * GPT_E,
                                           ln1b + l * GPT_E, xn);
    gemm_qkv_k<<<gQKV, 256, 0, stream>>>(xn, wq + oEE, wk + oEE, wv + oEE,
                                         qb, kb, vb);
    attn_k<<<gATT, 128, 0, stream>>>(qb, kb, vb, attb);
    gemm_k<true, false, true, true><<<gEE, 256, 0, stream>>>(
        attb, GPT_E, wproj + oEE, GPT_E, bproj + l * GPT_E, x, x, nullptr,
        GPT_E, GPT_E);
    layernorm_k<<<GPT_M, 256, 0, stream>>>(x, ln2g + l * GPT_E,
                                           ln2b + l * GPT_E, xn);
    gemm_k<true, true, false, false><<<gEF, 256, 0, stream>>>(
        xn, GPT_E, wfc + oEF, GPT_F4, bfc + l * GPT_F4, nullptr, nullptr, hb,
        GPT_F4, GPT_E);
    gemm_k<true, false, true, true><<<gEE, 256, 0, stream>>>(
        hb, GPT_F4, wpr2 + oEF, GPT_E, bpr2 + l * GPT_E, x, x, nullptr,
        GPT_E, GPT_F4);
  }

  layernorm_k<<<GPT_M, 256, 0, stream>>>(x, lnfg, lnfb, xn);
  gemm_k<true, false, false, true><<<gEV, 256, 0, stream>>>(
      xn, GPT_E, lmw, GPT_V, lmb, nullptr, (float*)d_out, nullptr,
      GPT_V, GPT_E);
}
